// EdgeClassNet_78434692759950
// MI455X (gfx1250) — matrix operand order repaired and checked by reading, not compiled, not run on hardware
//
#include <hip/hip_runtime.h>
#include <hip/hip_bf16.h>

// ---------- types ----------
typedef __attribute__((ext_vector_type(16))) __bf16 v16bf;
typedef __attribute__((ext_vector_type(8)))  float  v8f;
typedef unsigned int v4u __attribute__((ext_vector_type(4)));
typedef int          v8i __attribute__((ext_vector_type(8)));
typedef int          v4i __attribute__((ext_vector_type(4)));

union FragBF { v16bf v; uint4 q[2]; };

__device__ __forceinline__ unsigned short f2bf(float f) {
  union { float f; unsigned int u; } c; c.f = f;
  unsigned int u = c.u + 0x7FFFu + ((c.u >> 16) & 1u);   // round-to-nearest-even
  return (unsigned short)(u >> 16);
}
__device__ __forceinline__ float bf2f(unsigned short h) {
  union { unsigned int u; float f; } c; c.u = ((unsigned int)h) << 16;
  return c.f;
}

__device__ __forceinline__ void wait_asynccnt0() {
#if __has_builtin(__builtin_amdgcn_s_wait_asynccnt)
  __builtin_amdgcn_s_wait_asynccnt(0);
#else
  asm volatile("s_wait_asynccnt 0x0" ::: "memory");
#endif
}
__device__ __forceinline__ void wait_tensorcnt0() {
#if __has_builtin(__builtin_amdgcn_s_wait_tensorcnt)
  __builtin_amdgcn_s_wait_tensorcnt(0);
#else
  asm volatile("s_wait_tensorcnt 0x0" ::: "memory");
#endif
}

// async copy of 16B global -> LDS (tracked on ASYNCcnt, no VGPR round-trip)
__device__ __forceinline__ void async_g2l_b128(const void* gsrc, void* ldst) {
  unsigned int lptr = (unsigned int)(size_t)ldst;   // low 32 bits of generic ptr = LDS offset
  asm volatile("global_load_async_to_lds_b128 %0, %1, off"
               :: "v"(lptr), "v"(gsrc) : "memory");
}

// A fragment (16x32 bf16, M x K) from an LDS row-major tile.
// ISA layout: lane(0..15) row M=lane, halfs 0..7 = K[8kg..8kg+7], halfs 8..15 = K[16+8kg..]
__device__ __forceinline__ v16bf load_a_frag_lds(const unsigned short* lds, int mbase,
                                                 int kbase, int strideHalfs) {
  int lane = threadIdx.x & 31;
  int m = lane & 15, kg = lane >> 4;
  const unsigned short* p = lds + (mbase + m) * strideHalfs + kbase + 8 * kg;
  FragBF f;
  f.q[0] = *(const uint4*)(p);
  f.q[1] = *(const uint4*)(p + 16);
  return f.v;
}

// B fragment (32x16 bf16, K x N) from global weights stored N-major: Wt[n*Kl + k].
// ISA layout: col N = lane&15; halfs 0..7 = K[8kg..8kg+7], halfs 8..15 = K[16+8kg..] (kg = lane>>4).
__device__ __forceinline__ v16bf load_b_frag_g(const unsigned short* __restrict__ Wt,
                                               int nbase, int kbase, int Kl) {
  int lane = threadIdx.x & 31;
  int n = nbase + (lane & 15), kg = lane >> 4;
  const unsigned short* p = Wt + (size_t)n * Kl + kbase + 8 * kg;
  FragBF f;
  f.q[0] = *(const uint4*)(p);
  f.q[1] = *(const uint4*)(p + 16);
  return f.v;
}

// ---------- constants ----------
#define NN      65536
#define EE      262144
#define IN_DIM  128
#define HID     256
#define DNODE   256
#define K0PAD   288      // 272 padded to 9*32
#define NODE_MT 64
#define EDGE_MT 32
#define NEG_SLOPE 0.01f

// ---------- weight repack: f32 row-major -> bf16 N-major (transposed, padded) ----------
__global__ void pack_weights_kernel(const float* __restrict__ Wx,
                                    const float* __restrict__ W0,
                                    const float* __restrict__ Wmid,
                                    unsigned short* __restrict__ WxT,
                                    unsigned short* __restrict__ W0T,
                                    unsigned short* __restrict__ WmidT) {
  const int TOT_A = 128 * 256;            // WxT
  const int TOT_B = 256 * K0PAD;          // W0T
  const int TOT_C = 8 * 256 * 256;        // WmidT
  const int TOT   = TOT_A + TOT_B + TOT_C;
  for (int i = blockIdx.x * blockDim.x + threadIdx.x; i < TOT;
       i += gridDim.x * blockDim.x) {
    if (i < TOT_A) {
      int n = i >> 8, k = i & 255;
      WxT[n * 256 + k] = f2bf(Wx[k * 128 + n]);
    } else if (i < TOT_A + TOT_B) {
      int j = i - TOT_A;
      int n = j / K0PAD, k = j % K0PAD;
      W0T[n * K0PAD + k] = (k < 272) ? f2bf(W0[k * 256 + n]) : (unsigned short)0;
    } else {
      int j = i - (TOT_A + TOT_B);
      int l = j >> 16, r = j & 65535;
      int n = r >> 8, k = r & 255;
      WmidT[(size_t)l * 65536 + n * 256 + k] = f2bf(Wmid[(size_t)l * 65536 + k * 256 + n]);
    }
  }
}

// ---------- node MLP: h_node = leaky(x @ Wx + bx), out bf16 [N,128] ----------
__global__ __launch_bounds__(128) void node_mlp_kernel(const float* __restrict__ x,
                                                       const float* __restrict__ bx,
                                                       const unsigned short* __restrict__ WxT,
                                                       unsigned short* __restrict__ h_node) {
  __shared__ alignas(16) unsigned short lds_x[NODE_MT * DNODE];   // 32 KB
  const int tid = threadIdx.x;
  const int mb  = blockIdx.x * NODE_MT;

  // stage x tile as bf16
  for (int i = tid; i < NODE_MT * (DNODE / 4); i += 128) {
    int r = i >> 6, c4 = (i & 63) * 4;
    float4 v = *(const float4*)(x + (size_t)(mb + r) * DNODE + c4);
    unsigned short* d = lds_x + r * DNODE + c4;
    d[0] = f2bf(v.x); d[1] = f2bf(v.y); d[2] = f2bf(v.z); d[3] = f2bf(v.w);
  }
  __syncthreads();

  const int wave = tid >> 5, lane = tid & 31;
  const int mbase = wave * 16;
  const int ncol = lane & 15, kg = lane >> 4;

  v8f acc[8];
  #pragma unroll
  for (int t = 0; t < 8; ++t) {
    float b = bx[t * 16 + ncol];
    #pragma unroll
    for (int r = 0; r < 8; ++r) acc[t][r] = b;
  }
  #pragma unroll
  for (int ks = 0; ks < 8; ++ks) {          // K = 256
    v16bf a = load_a_frag_lds(lds_x, mbase, ks * 32, DNODE);
    #pragma unroll
    for (int g = 0; g < 2; ++g) {           // two groups of 4 n-tiles: batch loads, batch wmma
      v16bf bf[4];
      #pragma unroll
      for (int t = 0; t < 4; ++t)
        bf[t] = load_b_frag_g(WxT, (g * 4 + t) * 16, ks * 32, 256);
      #pragma unroll
      for (int t = 0; t < 4; ++t)
        acc[g * 4 + t] = __builtin_amdgcn_wmma_f32_16x16x32_bf16(
            false, a, false, bf[t], (short)0, acc[g * 4 + t], false, false);
    }
  }
  #pragma unroll
  for (int t = 0; t < 8; ++t)
    #pragma unroll
    for (int r = 0; r < 8; ++r) {
      float v = acc[t][r];
      v = (v >= 0.f) ? v : NEG_SLOPE * v;
      int m = mbase + r + 8 * kg;
      h_node[(size_t)(mb + m) * IN_DIM + t * 16 + ncol] = f2bf(v);
    }
}

// one wave computes a [16 x 64] strip of [32 x 256]: leaky(src @ Wt + bias) -> dst (bf16, stride 256)
__device__ __forceinline__ void wave_layer(const unsigned short* src, int sstride, int ksteps,
                                           const unsigned short* __restrict__ Wt, int Kl,
                                           const float* __restrict__ bias,
                                           unsigned short* dst) {
  const int tid = threadIdx.x;
  const int wave = tid >> 5, lane = tid & 31;
  const int mbase = (wave & 1) * 16;
  const int nbase = (wave >> 1) * 64;
  const int ncol = lane & 15, kg = lane >> 4;

  v8f acc[4];
  #pragma unroll
  for (int t = 0; t < 4; ++t) {
    float b = bias[nbase + t * 16 + ncol];
    #pragma unroll
    for (int r = 0; r < 8; ++r) acc[t][r] = b;
  }
  #pragma unroll
  for (int ks = 0; ks < ksteps; ++ks) {
    v16bf a = load_a_frag_lds(src, mbase, ks * 32, sstride);
    v16bf bf[4];                       // batch all B loads, then batch all wmma
    #pragma unroll
    for (int t = 0; t < 4; ++t)
      bf[t] = load_b_frag_g(Wt, nbase + t * 16, ks * 32, Kl);
    #pragma unroll
    for (int t = 0; t < 4; ++t)
      acc[t] = __builtin_amdgcn_wmma_f32_16x16x32_bf16(false, a, false, bf[t],
                                                       (short)0, acc[t], false, false);
  }
  #pragma unroll
  for (int t = 0; t < 4; ++t)
    #pragma unroll
    for (int r = 0; r < 8; ++r) {
      float v = acc[t][r];
      v = (v >= 0.f) ? v : NEG_SLOPE * v;
      dst[(mbase + r + 8 * kg) * HID + nbase + t * 16 + ncol] = f2bf(v);
    }
}

// ---------- fused edge pipeline: async gather + TDM attr tile -> 9 WMMA layers -> log_softmax ----
__global__ __launch_bounds__(256) void edge_mlp_kernel(const int* __restrict__ ei,
                                                       const float* __restrict__ edge_attr,
                                                       const unsigned short* __restrict__ h_node,
                                                       const unsigned short* __restrict__ W0T,
                                                       const float* __restrict__ b0,
                                                       const unsigned short* __restrict__ WmidT,
                                                       const float* __restrict__ b_mid,
                                                       const float* __restrict__ Wlast,
                                                       const float* __restrict__ blast,
                                                       float* __restrict__ out) {
  __shared__ alignas(16) unsigned short lds_ef[EDGE_MT * K0PAD];   // 18 KB
  __shared__ alignas(16) unsigned short lds_hA[EDGE_MT * HID];     // 16 KB
  __shared__ alignas(16) unsigned short lds_hB[EDGE_MT * HID];     // 16 KB
  __shared__ alignas(16) float lds_attr[EDGE_MT * 16];             // 2 KB TDM staging
  __shared__ float lds_wl[256 * 3];
  __shared__ float lds_logit[EDGE_MT * 3];

  const int tid = threadIdx.x;
  const int eb  = blockIdx.x * EDGE_MT;

  // ---- TDM: DMA the [32 x 16] f32 edge_attr tile into LDS (wave 0 issues; TENSORcnt) ----
#if __has_builtin(__builtin_amdgcn_tensor_load_to_lds)
  if ((tid >> 5) == 0) {
    unsigned long long ga = (unsigned long long)(const void*)(edge_attr + (size_t)eb * 16);
    unsigned int lad = (unsigned int)(size_t)(void*)lds_attr;
    v4u g0;
    g0.x = 1u;                                          // count=1 valid user descriptor
    g0.y = lad;                                         // lds_addr
    g0.z = (unsigned int)(ga & 0xFFFFFFFFu);            // global_addr[31:0]
    g0.w = (unsigned int)((ga >> 32) & 0x01FFFFFFu)     // global_addr[56:32]
         | (2u << 30);                                  // type = 2 ("image")
    v8i g1;
    g1[0] = 0x00020000;     // wg_mask=0, data_size=2 (4 bytes)
    g1[1] = 16 << 16;       // tensor_dim0 = 16
    g1[2] = 32 << 16;       // tensor_dim1 = 32
    g1[3] = 16 << 16;       // tile_dim0 = 16
    g1[4] = 32;             // tile_dim1 = 32, tile_dim2 = 0
    g1[5] = 16;             // tensor_dim0_stride = 16
    g1[6] = 0;
    g1[7] = 0;
    v4i z4 = {0, 0, 0, 0};  // groups 2/3 unused (2D tensor)
#if __clang_major__ >= 23
    v8i z8 = {0, 0, 0, 0, 0, 0, 0, 0};
    __builtin_amdgcn_tensor_load_to_lds(g0, g1, z4, z4, z8, 0);
#else
    __builtin_amdgcn_tensor_load_to_lds(g0, g1, z4, z4, 0);
#endif
  }
#else
  if (tid < EDGE_MT * 4) {   // fallback: plain vector loads
    ((float4*)lds_attr)[tid] = ((const float4*)(edge_attr + (size_t)eb * 16))[tid];
  }
#endif

  // ---- async gather of h_node[row]/h_node[col] straight into LDS (ASYNCcnt):
  // 32 rows x 2 node-parts x 16 chunks of 16B = 1024 items = exactly 4 full-EXEC sweeps
  for (int i = tid; i < EDGE_MT * 32; i += 256) {
    int e = i >> 5;
    int p = (i >> 4) & 1, c = i & 15;
    int node = ei[p * EE + eb + e];
    const unsigned short* g = h_node + (size_t)node * IN_DIM + c * 8;
    unsigned short* d = lds_ef + e * K0PAD + p * 128 + c * 8;
    async_g2l_b128(g, d);
  }

  // meanwhile: stage Wlast (f32, tiny)
  for (int i = tid; i < 768; i += 256) lds_wl[i] = Wlast[i];

  wait_tensorcnt0();          // wave 0 drains its TDM op (no-op for others)
  // wave 0 converts the attr tile f32 -> bf16 and writes the zero pad
  if (tid < EDGE_MT) {
    unsigned short* d = lds_ef + tid * K0PAD + 256;
    const float* s = lds_attr + tid * 16;
    #pragma unroll
    for (int j = 0; j < 16; ++j) d[j] = f2bf(s[j]);
    uint4 z; z.x = z.y = z.z = z.w = 0u;
    *(uint4*)(d + 16) = z;
    *(uint4*)(d + 24) = z;
  }

  wait_asynccnt0();           // each wave drains its own async copies
  __syncthreads();            // tile now visible to all waves

  // layer 0: ef(K=288) -> hA
  wave_layer(lds_ef, K0PAD, 9, W0T, K0PAD, b0, lds_hA);
  __syncthreads();

  // 8 mid layers, ping-pong hA <-> hB (unrolled so LDS addrspace stays inferable)
  #pragma unroll
  for (int l = 0; l < 8; ++l) {
    if (l < 7) __builtin_prefetch(WmidT + (size_t)(l + 1) * 65536, 0, 1);
    unsigned short* s = (l & 1) ? lds_hB : lds_hA;
    unsigned short* d = (l & 1) ? lds_hA : lds_hB;
    wave_layer(s, HID, 8, WmidT + (size_t)l * 65536, 256, b_mid + l * 256, d);
    __syncthreads();
  }
  // after 8 layers (l=7 wrote lds_hA) final activations are in lds_hA

  // last linear (256 -> 3) + log_softmax: tiny, VALU
  if (tid < EDGE_MT * 3) {
    int e = tid / 3, o = tid % 3;
    float s = blast[o];
    #pragma unroll 8
    for (int k = 0; k < HID; ++k)
      s += bf2f(lds_hA[e * HID + k]) * lds_wl[k * 3 + o];
    lds_logit[e * 3 + o] = s;
  }
  __syncthreads();
  if (tid < EDGE_MT) {
    float l0 = lds_logit[tid * 3 + 0];
    float l1 = lds_logit[tid * 3 + 1];
    float l2 = lds_logit[tid * 3 + 2];
    float m  = fmaxf(l0, fmaxf(l1, l2));
    float lse = m + __logf(__expf(l0 - m) + __expf(l1 - m) + __expf(l2 - m));
    size_t base = (size_t)(eb + tid) * 3;
    out[base + 0] = l0 - lse;
    out[base + 1] = l1 - lse;
    out[base + 2] = l2 - lse;
  }
}

// ---------- launch ----------
extern "C" void kernel_launch(void* const* d_in, const int* in_sizes, int n_in,
                              void* d_out, int out_size, void* d_ws, size_t ws_size,
                              hipStream_t stream) {
  (void)in_sizes; (void)n_in; (void)out_size; (void)ws_size;
  const float* x     = (const float*)d_in[0];
  const int*   ei    = (const int*)d_in[1];
  const float* ea    = (const float*)d_in[2];
  const float* Wx    = (const float*)d_in[3];
  const float* bx    = (const float*)d_in[4];
  const float* W0    = (const float*)d_in[5];
  const float* b0    = (const float*)d_in[6];
  const float* Wmid  = (const float*)d_in[7];
  const float* bmid  = (const float*)d_in[8];
  const float* Wlast = (const float*)d_in[9];
  const float* blast = (const float*)d_in[10];
  float* out = (float*)d_out;

  char* ws = (char*)d_ws;
  unsigned short* h_node = (unsigned short*)ws;                       // 65536*128*2 = 16 MB
  size_t off = (size_t)NN * IN_DIM * 2;
  unsigned short* WxT   = (unsigned short*)(ws + off); off += 128 * 256 * 2;
  unsigned short* W0T   = (unsigned short*)(ws + off); off += 256 * K0PAD * 2;
  unsigned short* WmidT = (unsigned short*)(ws + off);                // 8*256*256*2 = 1 MB

  pack_weights_kernel<<<1024, 256, 0, stream>>>(Wx, W0, Wmid, WxT, W0T, WmidT);
  node_mlp_kernel<<<NN / NODE_MT, 128, 0, stream>>>(x, bx, WxT, h_node);
  edge_mlp_kernel<<<EE / EDGE_MT, 256, 0, stream>>>(ei, ea, h_node, W0T, b0,
                                                    WmidT, bmid, Wlast, blast, out);
}
